// LabelGenerator_74887049773695
// MI455X (gfx1250) — 16-bit and 8-bit matrix operand order repaired and checked by reading, compiled, not run on hardware
//
#include <hip/hip_runtime.h>

typedef __attribute__((ext_vector_type(8)))  int      v8i;
typedef __attribute__((ext_vector_type(8)))  float    v8f;
typedef __attribute__((ext_vector_type(16))) _Float16 v16h;
typedef __attribute__((ext_vector_type(8)))  _Float16 v8h;
typedef __attribute__((ext_vector_type(4)))  float    v4f;
typedef __attribute__((ext_vector_type(4)))  int      v4i;

#define BATCH 32
#define H 768
#define W 768
#define WP 832
#define PADL 24
#define TILES_X (W / 16)
#define TILES_Y (H / 16)
#define TILES_PER_IMG (TILES_X * TILES_Y)
#define NTILES (BATCH * TILES_PER_IMG)
#define WAVES_PER_BLOCK 8

#define FP8_ONE 0x38u

__global__ __launch_bounds__(256) void pad_init(_Float16* __restrict__ V35,
                                                _Float16* __restrict__ V31) {
    const int PER_ROW = PADL + 40;
    int i = blockIdx.x * blockDim.x + threadIdx.x;
    int total = BATCH * H * PER_ROW;
    if (i >= total) return;
    int slot = i % PER_ROW;
    int br   = i / PER_ROW;
    int cs = (slot < PADL) ? slot : (WP - 40 + (slot - PADL));
    size_t off = (size_t)br * WP + cs;
    V35[off] = (_Float16)0.0f;
    V31[off] = (_Float16)0.0f;
}

__global__ __launch_bounds__(256) void pass1_vert(const float* __restrict__ X,
                                                  _Float16* __restrict__ V35,
                                                  _Float16* __restrict__ V31) {
    const int lane = threadIdx.x & 31;
    const int wave = threadIdx.x >> 5;
    const int tile = blockIdx.x * WAVES_PER_BLOCK + wave;
    const int b  = tile / TILES_PER_IMG;
    const int tr = tile % TILES_PER_IMG;
    const int ty = tr / TILES_X, tx = tr % TILES_X;
    const int r0 = ty * 16, c0 = tx * 16;
    const int lo = lane & 15;
    const int hi = (lane >> 4) & 1;

    const float* __restrict__ Xb = X + (size_t)b * H * W;

    v8i A;
#pragma unroll
    for (int v = 0; v < 8; ++v) {
        unsigned d = 0u;
#pragma unroll
        for (int by = 0; by < 4; ++by) {
            int j   = 32 * (v >> 2) + 16 * hi + 4 * (v & 3) + by;
            int row = r0 - 17 + j;
            float x = 0.0f;
            if ((unsigned)row < (unsigned)H)
                x = Xb[(size_t)row * W + (c0 + lo)];
            d |= ((x > 0.5f) ? FP8_ONE : 0u) << (8 * by);
        }
        A[v] = (int)d;
    }

    v8i B35, B31;
#pragma unroll
    for (int v = 0; v < 8; ++v) {
        unsigned d35 = 0u, d31 = 0u;
#pragma unroll
        for (int by = 0; by < 4; ++by) {
            int j = 32 * (v >> 2) + 16 * hi + 4 * (v & 3) + by;
            int m = lo;
            if (j >= m     && j <= m + 34) d35 |= FP8_ONE << (8 * by);
            if (j >= m + 2 && j <= m + 32) d31 |= FP8_ONE << (8 * by);
        }
        B35[v] = (int)d35;
        B31[v] = (int)d31;
    }

    v8f zero = {};
    v8f c35 = __builtin_amdgcn_wmma_f32_16x16x64_fp8_fp8(A, B35, (short)0, zero, false, false);
    v8f c31 = __builtin_amdgcn_wmma_f32_16x16x64_fp8_fp8(A, B31, (short)0, zero, false, false);

    const int r = r0 + lo;
    size_t base = ((size_t)b * H + r) * WP + (size_t)(PADL + c0 + 8 * hi);
    v8h s35, s31;
#pragma unroll
    for (int g = 0; g < 8; ++g) {
        s35[g] = (_Float16)c35[g];
        s31[g] = (_Float16)c31[g];
    }
    *(v8h*)(V35 + base) = s35;
    *(v8h*)(V31 + base) = s31;
}

__global__ __launch_bounds__(256) void pass2_horiz(const float* __restrict__ X,
                                                   const _Float16* __restrict__ V35,
                                                   const _Float16* __restrict__ V31,
                                                   float* __restrict__ rsm,
                                                   int*   __restrict__ pfm) {
    const int lane = threadIdx.x & 31;
    const int wave = threadIdx.x >> 5;
    const int tile = blockIdx.x * WAVES_PER_BLOCK + wave;
    const int b  = tile / TILES_PER_IMG;
    const int tr = tile % TILES_PER_IMG;
    const int ty = tr / TILES_X, tx = tr % TILES_X;
    const int r0 = ty * 16, c0 = tx * 16;
    const int lo = lane & 15;
    const int hi = (lane >> 4) & 1;

    v16h A35[2], A31[2];
#pragma unroll
    for (int ch = 0; ch < 2; ++ch) {
#pragma unroll
        for (int h = 0; h < 16; ++h) {
            int K = h + 16 * hi;
            int j = 32 * ch + K;
            int n = lo;
            A35[ch][h] = (j >= n + 7 && j <= n + 41) ? (_Float16)1.0f : (_Float16)0.0f;
            A31[ch][h] = (j >= n + 9 && j <= n + 39) ? (_Float16)1.0f : (_Float16)0.0f;
        }
    }

    const int r = r0 + lo;
    size_t vrow = ((size_t)b * H + r) * WP;
    v16h B35[2], B31[2];
#pragma unroll
    for (int ch = 0; ch < 2; ++ch) {
        size_t off = vrow + (size_t)(c0 + 32 * ch + 16 * hi);
        B35[ch] = *(const v16h*)(V35 + off);
        B31[ch] = *(const v16h*)(V31 + off);
    }

    v8f zero = {};
    v8f s35 = __builtin_amdgcn_wmma_f32_16x16x32_f16(false, A35[0], false, B35[0], (short)0, zero, false, false);
    s35     = __builtin_amdgcn_wmma_f32_16x16x32_f16(false, A35[1], false, B35[1], (short)0, s35,  false, false);
    v8f s31 = __builtin_amdgcn_wmma_f32_16x16x32_f16(false, A31[0], false, B31[0], (short)0, zero, false, false);
    s31     = __builtin_amdgcn_wmma_f32_16x16x32_f16(false, A31[1], false, B31[1], (short)0, s31,  false, false);

    size_t obase = ((size_t)b * H + r) * W + (size_t)(c0 + 8 * hi);
    v4f x0 = *(const v4f*)(X + obase);
    v4f x1 = *(const v4f*)(X + obase + 4);

    const float inv = 1.0f / 1225.0f;
    v4f o0, o1;
    v4i p0, p1;
#pragma unroll
    for (int g = 0; g < 4; ++g) {
        o0[g] = s35[g] * inv;
        o1[g] = s35[g + 4] * inv;
        p0[g] = (x0[g] > 0.5f) ? 1 : ((s31[g]     > 0.5f) ? 0 : 2);
        p1[g] = (x1[g] > 0.5f) ? 1 : ((s31[g + 4] > 0.5f) ? 0 : 2);
    }
    *(v4f*)(rsm + obase)     = o0;
    *(v4f*)(rsm + obase + 4) = o1;
    *(v4i*)(pfm + obase)     = p0;
    *(v4i*)(pfm + obase + 4) = p1;
}

extern "C" void kernel_launch(void* const* d_in, const int* in_sizes, int n_in,
                              void* d_out, int out_size, void* d_ws, size_t ws_size,
                              hipStream_t stream) {
    const float* X = (const float*)d_in[0];
    const size_t NTOT = (size_t)BATCH * H * W;

    float* rsm = (float*)d_out;
    int*   pfm = (int*)((float*)d_out + NTOT);

    _Float16* V35 = (_Float16*)d_ws;
    _Float16* V31 = V35 + (size_t)BATCH * H * WP;

    {
        int total = BATCH * H * (PADL + 40);
        pad_init<<<(total + 255) / 256, 256, 0, stream>>>(V35, V31);
    }
    pass1_vert <<<NTILES / WAVES_PER_BLOCK, 32 * WAVES_PER_BLOCK, 0, stream>>>(X, V35, V31);
    pass2_horiz<<<NTILES / WAVES_PER_BLOCK, 32 * WAVES_PER_BLOCK, 0, stream>>>(X, V35, V31, rsm, pfm);
}
